// MirasModel_35811437314289
// MI455X (gfx1250) — matrix operand order repaired and checked by reading, not compiled, not run on hardware
//
#include <hip/hip_runtime.h>
#include <hip/hip_bf16.h>
#include <cmath>

// ---------------- types ----------------
typedef __attribute__((ext_vector_type(16))) __bf16 v16bf;
typedef __attribute__((ext_vector_type(8)))  __bf16 v8bf;
typedef __attribute__((ext_vector_type(8)))  float  v8f;

union AB16 { v16bf v; v8bf h[2]; };

#define TT 512
#define CH 64
#define NCHUNK 8
#define DM 3136
#define DH 512

// ---------------- helpers ----------------
static __device__ inline unsigned short f2bf(float f) {
    unsigned int u = __float_as_uint(f);
    unsigned int r = u + 0x7FFFu + ((u >> 16) & 1u);
    return (unsigned short)(r >> 16);
}

static __device__ inline float gelu_tanh(float x) {
    const float c = 0.7978845608028654f;
    float u = c * (x + 0.044715f * x * x * x);
    float t = tanhf(u);
    return 0.5f * x * (1.0f + t);
}

static __device__ inline float gelu_tanh_grad(float x) {
    const float c = 0.7978845608028654f;
    float x2 = x * x;
    float u = c * (x + 0.044715f * x2 * x);
    float t = tanhf(u);
    float sech2 = 1.0f - t * t;
    return 0.5f * (1.0f + t) + 0.5f * x * sech2 * c * (1.0f + 3.0f * 0.044715f * x2);
}

// ---------------- conv3x3 + RMSNorm over C=4, emit K(bf16), K^T(bf16 per chunk), V(f32) ----------------
__global__ void conv_rms_kernel(const float* __restrict__ x,
                                const float* __restrict__ wk, const float* __restrict__ bk,
                                const float* __restrict__ wv, const float* __restrict__ bv,
                                const float* __restrict__ sk, const float* __restrict__ sv,
                                unsigned short* __restrict__ Kb,
                                unsigned short* __restrict__ Ktb,
                                float* __restrict__ Vf) {
    int idx = blockIdx.x * blockDim.x + threadIdx.x;
    if (idx >= TT * 28 * 28) return;
    int t = idx / 784;
    int hw = idx % 784;
    int h = hw / 28, wc = hw % 28;
    float ak[4], av[4];
#pragma unroll
    for (int c = 0; c < 4; ++c) { ak[c] = bk[c]; av[c] = bv[c]; }
    for (int kh = 0; kh < 3; ++kh) {
        int hy = h + kh - 1; if (hy < 0 || hy >= 28) continue;
        for (int kw = 0; kw < 3; ++kw) {
            int wx = wc + kw - 1; if (wx < 0 || wx >= 28) continue;
#pragma unroll
            for (int ci = 0; ci < 4; ++ci) {
                float xi = x[((t * 4 + ci) * 28 + hy) * 28 + wx];
                int base = ((kh * 3 + kw) * 4 + ci) * 4;
#pragma unroll
                for (int co = 0; co < 4; ++co) {
                    ak[co] += xi * wk[base + co];
                    av[co] += xi * wv[base + co];
                }
            }
        }
    }
    float mk = 0.f, mv = 0.f;
#pragma unroll
    for (int c = 0; c < 4; ++c) { mk += ak[c] * ak[c]; mv += av[c] * av[c]; }
    float ik = rsqrtf(mk * 0.25f + 1e-6f);
    float iv = rsqrtf(mv * 0.25f + 1e-6f);
    int chunk = t >> 6, ti = t & 63;
#pragma unroll
    for (int c = 0; c < 4; ++c) {
        int d = hw * 4 + c;
        float kf = ak[c] * ik * sk[c];
        float vf = av[c] * iv * sv[c];
        Kb[(size_t)t * DM + d] = f2bf(kf);
        Ktb[((size_t)chunk * DM + d) * CH + ti] = f2bf(kf);
        Vf[(size_t)t * DM + d] = vf;
    }
}

// ---------------- copy masters + initial bf16 casts (W1t, W2 row, W2t) ----------------
__global__ void init_params_kernel(const float* __restrict__ W1, const float* __restrict__ b1,
                                   const float* __restrict__ W2, const float* __restrict__ b2,
                                   float* __restrict__ W1m, float* __restrict__ W2m,
                                   float* __restrict__ b1m, float* __restrict__ b2m,
                                   unsigned short* __restrict__ W1t,
                                   unsigned short* __restrict__ W2b,
                                   unsigned short* __restrict__ W2t) {
    int idx = blockIdx.x * blockDim.x + threadIdx.x;
    if (idx < DM * DH) {
        // W1 is [DM x DH] row-major
        int m = idx / DH, n = idx % DH;
        float w1 = W1[idx];
        W1m[idx] = w1;
        W1t[(size_t)n * DM + m] = f2bf(w1);
        // W2 is [DH x DM] row-major; same flat count
        int j = idx / DM, d = idx % DM;
        float w2 = W2[idx];
        W2m[idx] = w2;
        W2b[idx] = f2bf(w2);
        W2t[(size_t)d * DH + j] = f2bf(w2);
    }
    if (idx < DH) b1m[idx] = b1[idx];
    if (idx < DM) b2m[idx] = b2[idx];
}

// ---------------- per-chunk bias update: b -= w * colsum ----------------
__global__ void bias_update_kernel(const float* __restrict__ dHf, const float* __restrict__ Rf,
                                   float* __restrict__ b1m, float* __restrict__ b2m, float wsc) {
    int idx = blockIdx.x * blockDim.x + threadIdx.x;
    if (idx < DH) {
        float s = 0.f;
        for (int i = 0; i < CH; ++i) s += dHf[(size_t)i * DH + idx];
        b1m[idx] -= wsc * s;
    } else if (idx < DH + DM) {
        int j = idx - DH;
        float s = 0.f;
        for (int i = 0; i < CH; ++i) s += Rf[(size_t)i * DM + j];
        b2m[j] -= wsc * s;
    }
}

// ---------------- generic NT bf16 WMMA GEMM, one 16x16 tile per wave, fused epilogues ----------------
// EPI 1: h=acc+bias; of?=h ; ob=gelu(h) bf16 ; obt?=gelu(h)^T bf16
// EPI 2: r=2*((acc+bias)-aux); of=r ; ob=r bf16 ; obt=r^T bf16
// EPI 3: dh=acc*gelu'(aux); of=dh ; obt=dh^T bf16
// EPI 4: of (master, RMW) -= wsc*acc; ob?=new bf16 ; obt=new^T bf16
// EPI 5: of = acc + bias
template <int EPI>
__global__ __launch_bounds__(256) void gemm_bf16_wmma(
    const unsigned short* __restrict__ A, const unsigned short* __restrict__ Bt,
    int ntn, int ntotal, int lda, int ldb, int K,
    const float* __restrict__ bias, const float* __restrict__ aux,
    float* __restrict__ of, unsigned short* __restrict__ ob, unsigned short* __restrict__ obt,
    float wsc, int ldo, int ldt) {
    int wave = blockIdx.x * 8 + (threadIdx.x >> 5);
    if (wave >= ntotal) return;                 // uniform per wave: EXEC stays all-ones
    int tm = wave / ntn, tn = wave % ntn;
    int lane = threadIdx.x & 31;
    int half = lane >> 4, l16 = lane & 15;

    // A 16x32 bf16 fragment: lane half 0 -> K 0..15, half 1 -> K 16..31
    const unsigned short* ap = A + (size_t)(tm * 16 + l16) * lda + half * 16;
    // B 32x16 fragment from Bt (row-major [N x K]): lane half 0 -> K 0..15, half 1 -> K 16..31
    const unsigned short* bp = Bt + (size_t)(tn * 16 + l16) * ldb + half * 16;

    v8f acc = {0.f, 0.f, 0.f, 0.f, 0.f, 0.f, 0.f, 0.f};
    for (int k0 = 0; k0 < K; k0 += 32) {
        AB16 af;
        af.h[0] = *(const v8bf*)(ap);           // 16B load
        af.h[1] = *(const v8bf*)(ap + 8);       // 16B load
        v16bf bf = *(const v16bf*)(bp);         // 32B load
        acc = __builtin_amdgcn_wmma_f32_16x16x32_bf16(
            false, af.v, false, bf, (short)0, acc, false, false);
        ap += 32;
        bp += 32;
    }

    int n = tn * 16 + l16;
    int mbase = tm * 16 + half * 8;             // C layout: VGPR i -> M = i + 8*half, N = lane&15
#pragma unroll
    for (int i = 0; i < 8; ++i) {
        int m = mbase + i;
        float a = acc[i];
        if (EPI == 1) {
            float hv = a + bias[n];
            if (of) of[(size_t)m * ldo + n] = hv;
            float g = gelu_tanh(hv);
            ob[(size_t)m * ldo + n] = f2bf(g);
            if (obt) obt[(size_t)n * ldt + m] = f2bf(g);
        } else if (EPI == 2) {
            float r = 2.0f * ((a + bias[n]) - aux[(size_t)m * ldo + n]);
            of[(size_t)m * ldo + n] = r;
            ob[(size_t)m * ldo + n] = f2bf(r);
            obt[(size_t)n * ldt + m] = f2bf(r);
        } else if (EPI == 3) {
            float dh = a * gelu_tanh_grad(aux[(size_t)m * ldo + n]);
            of[(size_t)m * ldo + n] = dh;
            obt[(size_t)n * ldt + m] = f2bf(dh);
        } else if (EPI == 4) {
            size_t o = (size_t)m * ldo + n;
            float nv = of[o] - wsc * a;
            of[o] = nv;
            if (ob) ob[o] = f2bf(nv);
            obt[(size_t)n * ldt + m] = f2bf(nv);
        } else {
            of[(size_t)m * ldo + n] = a + bias[n];
        }
    }
}

// ---------------- host-side dispatch ----------------
static void launch_gemm(int epi, hipStream_t stream,
                        const unsigned short* A, const unsigned short* Bt,
                        int M, int N, int K, int lda, int ldb,
                        const float* bias, const float* aux,
                        float* of, unsigned short* ob, unsigned short* obt,
                        float wsc, int ldo, int ldt) {
    int ntn = N / 16;
    int ntotal = (M / 16) * ntn;
    dim3 grid((ntotal + 7) / 8), block(256);
    switch (epi) {
    case 1: gemm_bf16_wmma<1><<<grid, block, 0, stream>>>(A, Bt, ntn, ntotal, lda, ldb, K, bias, aux, of, ob, obt, wsc, ldo, ldt); break;
    case 2: gemm_bf16_wmma<2><<<grid, block, 0, stream>>>(A, Bt, ntn, ntotal, lda, ldb, K, bias, aux, of, ob, obt, wsc, ldo, ldt); break;
    case 3: gemm_bf16_wmma<3><<<grid, block, 0, stream>>>(A, Bt, ntn, ntotal, lda, ldb, K, bias, aux, of, ob, obt, wsc, ldo, ldt); break;
    case 4: gemm_bf16_wmma<4><<<grid, block, 0, stream>>>(A, Bt, ntn, ntotal, lda, ldb, K, bias, aux, of, ob, obt, wsc, ldo, ldt); break;
    default: gemm_bf16_wmma<5><<<grid, block, 0, stream>>>(A, Bt, ntn, ntotal, lda, ldb, K, bias, aux, of, ob, obt, wsc, ldo, ldt); break;
    }
}

extern "C" void kernel_launch(void* const* d_in, const int* in_sizes, int n_in,
                              void* d_out, int out_size, void* d_ws, size_t ws_size,
                              hipStream_t stream) {
    (void)in_sizes; (void)n_in; (void)out_size; (void)ws_size;
    const float* x       = (const float*)d_in[0];
    const float* conv_kw = (const float*)d_in[1];
    const float* conv_kb = (const float*)d_in[2];
    const float* conv_vw = (const float*)d_in[3];
    const float* conv_vb = (const float*)d_in[4];
    const float* rms_ks  = (const float*)d_in[5];
    const float* rms_vs  = (const float*)d_in[6];
    const float* W1      = (const float*)d_in[7];
    const float* b1      = (const float*)d_in[8];
    const float* W2      = (const float*)d_in[9];
    const float* b2      = (const float*)d_in[10];
    float* Y = (float*)d_out;

    // bump allocator over workspace (256B aligned chunks)
    char* w = (char*)d_ws;
    auto alloc = [&](size_t bytes) -> void* {
        void* p = (void*)w;
        w += (bytes + 255) & ~(size_t)255;
        return p;
    };
    unsigned short* Kb   = (unsigned short*)alloc((size_t)TT * DM * 2);       // K bf16 [T x DM]
    unsigned short* Ktb  = (unsigned short*)alloc((size_t)NCHUNK * DM * CH * 2); // K^T bf16 per chunk [DM x 64]
    float*          Vf   = (float*)alloc((size_t)TT * DM * 4);                // V f32
    float*          W1m  = (float*)alloc((size_t)DM * DH * 4);                // masters
    float*          W2m  = (float*)alloc((size_t)DH * DM * 4);
    float*          b1m  = (float*)alloc((size_t)DH * 4);
    float*          b2m  = (float*)alloc((size_t)DM * 4);
    unsigned short* W1t  = (unsigned short*)alloc((size_t)DH * DM * 2);       // W1^T bf16 [DH x DM]
    unsigned short* W2b  = (unsigned short*)alloc((size_t)DH * DM * 2);       // W2 bf16 [DH x DM]
    unsigned short* W2t  = (unsigned short*)alloc((size_t)DM * DH * 2);       // W2^T bf16 [DM x DH]
    float*          Hf   = (float*)alloc((size_t)CH * DH * 4);
    unsigned short* Gb   = (unsigned short*)alloc((size_t)CH * DH * 2);
    unsigned short* Gtb  = (unsigned short*)alloc((size_t)DH * CH * 2);
    float*          Rf   = (float*)alloc((size_t)CH * DM * 4);
    unsigned short* Rb   = (unsigned short*)alloc((size_t)CH * DM * 2);
    unsigned short* Rtb  = (unsigned short*)alloc((size_t)DM * CH * 2);
    float*          dHf  = (float*)alloc((size_t)CH * DH * 4);
    unsigned short* dHtb = (unsigned short*)alloc((size_t)DH * CH * 2);
    unsigned short* G2b  = (unsigned short*)alloc((size_t)CH * DH * 2);

    const float wsc = (float)(0.1 * std::pow(0.9, 63.0)); // constant per-token weight

    // preprocessing
    conv_rms_kernel<<<(TT * 784 + 255) / 256, 256, 0, stream>>>(
        x, conv_kw, conv_kb, conv_vw, conv_vb, rms_ks, rms_vs, Kb, Ktb, Vf);
    init_params_kernel<<<(DM * DH + 255) / 256, 256, 0, stream>>>(
        W1, b1, W2, b2, W1m, W2m, b1m, b2m, W1t, W2b, W2t);

    for (int ch = 0; ch < NCHUNK; ++ch) {
        const unsigned short* Kc  = Kb + (size_t)ch * CH * DM;
        const unsigned short* Ktc = Ktb + (size_t)ch * DM * CH;
        const float*          Vc  = Vf + (size_t)ch * CH * DM;
        float*                Yc  = Y + (size_t)ch * CH * DM;

        // 1) H = K@W1 + b1 ; G = gelu(H) (+ G^T)
        launch_gemm(1, stream, Kc, W1t, CH, DH, DM, DM, DM, b1m, nullptr, Hf, Gb, Gtb, 0.f, DH, CH);
        // 2) R = 2*(G@W2 + b2 - V) (+ R bf16, R^T)
        launch_gemm(2, stream, Gb, W2t, CH, DM, DH, DH, DH, b2m, Vc, Rf, Rb, Rtb, 0.f, DM, CH);
        // 3) dH = (R@W2^T) * gelu'(H) (+ dH^T)
        launch_gemm(3, stream, Rb, W2b, CH, DH, DM, DM, DM, nullptr, Hf, dHf, nullptr, dHtb, 0.f, DH, CH);
        // 4) b1 -= w*colsum(dH) ; b2 -= w*colsum(R)
        bias_update_kernel<<<(DH + DM + 255) / 256, 256, 0, stream>>>(dHf, Rf, b1m, b2m, wsc);
        // 5) W1 -= w * K^T @ dH (re-emit W1^T bf16)
        launch_gemm(4, stream, Ktc, dHtb, DM, DH, CH, CH, CH, nullptr, nullptr, W1m, nullptr, W1t, wsc, DH, DM);
        // 6) W2 -= w * G^T @ R (re-emit W2 bf16 + W2^T bf16)
        launch_gemm(4, stream, Gtb, Rtb, DH, DM, CH, CH, CH, nullptr, nullptr, W2m, W2b, W2t, wsc, DM, DH);
        // 7) G2 = gelu(K@W1' + b1')
        launch_gemm(1, stream, Kc, W1t, CH, DH, DM, DM, DM, b1m, nullptr, nullptr, G2b, nullptr, 0.f, DH, CH);
        // 8) Y = G2@W2' + b2'
        launch_gemm(5, stream, G2b, W2t, CH, DM, DH, DH, DH, b2m, nullptr, Yc, nullptr, nullptr, 0.f, DM, 0);
    }
}
